// NeighborhoodAttentionSplitHead_73675868995692
// MI455X (gfx1250) — matrix operand order repaired and checked by reading, not compiled, not run on hardware
//
#include <hip/hip_runtime.h>
#include <hip/hip_bf16.h>

#define DEVINL static __device__ __forceinline__

typedef __attribute__((ext_vector_type(16))) __bf16         bf16x16;
typedef __attribute__((ext_vector_type(8)))  float          f32x8;
typedef __attribute__((ext_vector_type(16))) unsigned short u16x16;
typedef __attribute__((ext_vector_type(4)))  unsigned int   u32x4;
typedef __attribute__((ext_vector_type(4)))  float          f32x4;

union FragB16 {
    bf16x16 bf;
    u16x16  us;
    u32x4   q4[2];
};

DEVINL unsigned short f2bf(float f) {
    unsigned u = __float_as_uint(f);
    u += 0x7fffu + ((u >> 16) & 1u);            // round-to-nearest-even
    return (unsigned short)(u >> 16);
}
DEVINL unsigned f2bf_pk(float lo, float hi) {
    return (unsigned)f2bf(lo) | ((unsigned)f2bf(hi) << 16);
}
DEVINL int imin(int a, int b) { return a < b ? a : b; }
DEVINL int imax(int a, int b) { return a > b ? a : b; }

DEVINL f32x8 wmma_bf16(const FragB16& a, const FragB16& b, f32x8 c) {
    return __builtin_amdgcn_wmma_f32_16x16x32_bf16(false, a.bf, false, b.bf,
                                                   (short)0, c, false, false);
}

// ---------------------------------------------------------------------------
// Kernel 0: bulk fp32 -> bf16 conversion (8 elements/thread, b128 in/out).
// n must be a multiple of 8.
// ---------------------------------------------------------------------------
__global__ __launch_bounds__(256) void f32_to_bf16_kernel(
    const float* __restrict__ src, unsigned short* __restrict__ dst, int n)
{
    const int i = (blockIdx.x * 256 + threadIdx.x) * 8;
    if (i >= n) return;
    const f32x4 a = *(const f32x4*)(src + i);
    const f32x4 b = *(const f32x4*)(src + i + 4);
    u32x4 o;
    o[0] = f2bf_pk(a[0], a[1]);
    o[1] = f2bf_pk(a[2], a[3]);
    o[2] = f2bf_pk(b[0], b[1]);
    o[3] = f2bf_pk(b[2], b[3]);
    *(u32x4*)(dst + i) = o;
}

// ---------------------------------------------------------------------------
// Kernel 1: qkv = x @ w_qkv^T + b_qkv  (M=32768, N=768, K=256), bf16 WMMA.
// 32x64 tile per wave (8 wmma per K-chunk). bf16 pre-converted operands.
// Epilogue: q (scaled) -> qb[bh][pix][32], k -> kb[bh][pix][32],
//           v -> vtb[bh][d][pix]   (transposed for aligned PV fragments)
// ---------------------------------------------------------------------------
__global__ __launch_bounds__(256) void qkv_wmma_kernel(
    const unsigned short* __restrict__ xbf, const unsigned short* __restrict__ wqbf,
    const float* __restrict__ bq,
    unsigned short* __restrict__ qb, unsigned short* __restrict__ kb,
    unsigned short* __restrict__ vtb)
{
    const int lane = threadIdx.x & 31;
    const int warp = threadIdx.x >> 5;
    const int wid  = blockIdx.x * 8 + warp;
    const int m0   = (wid / 12) * 32;
    const int n0   = (wid % 12) * 64;
    const int l15  = lane & 15;
    const int hi   = lane >> 4;

    const unsigned short* arow0 = xbf + (size_t)(m0 + l15) * 256;
    const unsigned short* arow1 = xbf + (size_t)(m0 + 16 + l15) * 256;

    f32x8 acc[2][4];
    #pragma unroll
    for (int mi = 0; mi < 2; ++mi)
        #pragma unroll
        for (int nt = 0; nt < 4; ++nt)
            #pragma unroll
            for (int j = 0; j < 8; ++j) acc[mi][nt][j] = 0.f;

    #pragma unroll
    for (int kc = 0; kc < 8; ++kc) {
        FragB16 a0, a1;
        a0.q4[0] = *(const u32x4*)(arow0 + kc * 32 + hi * 8);
        a0.q4[1] = *(const u32x4*)(arow0 + kc * 32 + 16 + hi * 8);
        a1.q4[0] = *(const u32x4*)(arow1 + kc * 32 + hi * 8);
        a1.q4[1] = *(const u32x4*)(arow1 + kc * 32 + 16 + hi * 8);
        #pragma unroll
        for (int nt = 0; nt < 4; ++nt) {
            const int nn = n0 + nt * 16 + l15;
            const unsigned short* wrow = wqbf + (size_t)nn * 256 + kc * 32 + hi * 8;
            FragB16 bfr;
            bfr.q4[0] = *(const u32x4*)(wrow);
            bfr.q4[1] = *(const u32x4*)(wrow + 16);
            acc[0][nt] = wmma_bf16(a0, bfr, acc[0][nt]);
            acc[1][nt] = wmma_bf16(a1, bfr, acc[1][nt]);
        }
    }

    #pragma unroll
    for (int nt = 0; nt < 4; ++nt) {
        const int nn = n0 + nt * 16 + l15;
        const float bias = bq[nn];
        #pragma unroll
        for (int mi = 0; mi < 2; ++mi) {
            #pragma unroll
            for (int j = 0; j < 8; ++j) {
                const int row = m0 + mi * 16 + j + hi * 8;
                const int bb  = row >> 12;
                const int pix = row & 4095;
                const float v = acc[mi][nt][j] + bias;
                if (nn < 256) {
                    const int head = nn >> 5, d = nn & 31;
                    qb[((size_t)(bb * 8 + head) * 4096 + pix) * 32 + d] =
                        f2bf(v * 0.17677669529663687f);   // * 32^-0.5
                } else if (nn < 512) {
                    const int n2 = nn - 256, head = n2 >> 5, d = n2 & 31;
                    kb[((size_t)(bb * 8 + head) * 4096 + pix) * 32 + d] = f2bf(v);
                } else {
                    const int n3 = nn - 512, head = n3 >> 5, d = n3 & 31;
                    vtb[((size_t)(bb * 8 + head) * 32 + d) * 4096 + pix] = f2bf(v);
                }
            }
        }
    }
}

// ---------------------------------------------------------------------------
// Kernels 2/3: neighborhood attention, one wave per 4x4 query tile.
//  KS=7 : union 16x16 keys, NK=256, 4 waves/block (64KB LDS)
//  KS=13: union 16x32 keys, NK=512, 2 waves/block (64KB LDS)
// Union anchor aligned so all K/V fragment loads are 16B-aligned b128.
// ---------------------------------------------------------------------------
template <int KS, int NU, int WAVES>
__global__ __launch_bounds__(WAVES * 32) void natt_wmma_kernel(
    const unsigned short* __restrict__ qb, const unsigned short* __restrict__ kb,
    const unsigned short* __restrict__ vtb, const float* __restrict__ rpb,
    unsigned short* __restrict__ attn, int head_base)
{
    constexpr int NH  = KS / 2;
    constexpr int UWS = (NU == 1) ? 4 : 5;
    constexpr int UW  = 1 << UWS;          // union width (keys)
    constexpr int NK  = 16 * UW;           // total union keys (256 / 512)
    constexpr int RPW = 2 * KS - 1;

    __shared__ float smem[WAVES][16 * NK];

    const int lane = threadIdx.x & 31;
    const int warp = threadIdx.x >> 5;
    float* sw = smem[warp];

    const int wid    = blockIdx.x * WAVES + warp;
    const int b      = wid >> 10;
    const int r1     = wid & 1023;
    const int head_l = r1 >> 8;
    const int t2     = r1 & 255;
    const int qh0    = (t2 >> 4) * 4;
    const int qw0    = (t2 & 15) * 4;
    const int head   = head_base + head_l;
    const int bh     = b * 8 + head;
    const size_t pbase = (size_t)bh * 4096;

    // union anchor (row: plain clamp; col: aligned for b128 V loads)
    int uh = imin(imin(imax(qh0 - NH, 0), 64 - KS), 48);
    int uw = imin(imax(qw0 - NH, 0), 64 - KS);
    uw = (NU == 1) ? imin(uw & ~7, 48) : imin(uw & ~15, 32);

    const int l15 = lane & 15;
    const int hi  = lane >> 4;

    // ---- scores S[16 q, NK keys] = Q * K^T ----
    {
        const int qh = qh0 + (l15 >> 2);
        const int qw = qw0 + (l15 & 3);
        const unsigned short* qrow = qb + (pbase + qh * 64 + qw) * 32;
        FragB16 aq;
        aq.q4[0] = *(const u32x4*)(qrow + hi * 8);
        aq.q4[1] = *(const u32x4*)(qrow + 16 + hi * 8);

        #pragma unroll
        for (int t = 0; t < NK / 16; ++t) {
            const int ky  = t / NU;
            const int kxb = (t % NU) * 16;
            const unsigned short* krow =
                kb + (pbase + (size_t)(uh + ky) * 64 + (uw + kxb + l15)) * 32;
            FragB16 bk;
            bk.q4[0] = *(const u32x4*)(krow + hi * 8);
            bk.q4[1] = *(const u32x4*)(krow + hi * 8 + 16);
            f32x8 s;
            #pragma unroll
            for (int j = 0; j < 8; ++j) s[j] = 0.f;
            s = wmma_bf16(aq, bk, s);
            #pragma unroll
            for (int j = 0; j < 8; ++j)
                sw[(j + hi * 8) * NK + t * 16 + l15] = s[j];
        }
    }
    __syncthreads();

    // ---- mask + relative bias + softmax (lane q handles query row q) ----
    if (lane < 16) {
        const int q = lane;
        float* prow = sw + q * NK;
        const int qh = qh0 + (q >> 2);
        const int qw = qw0 + (q & 3);
        const int sh = imin(imax(qh - NH, 0), 64 - KS);
        const int sv = imin(imax(qw - NH, 0), 64 - KS);
        const int ph = NH + imax(NH - qh, 0) + ((qh + NH >= 64) ? (64 - qh - 1 - NH) : 0);
        const int pw = NH + imax(NH - qw, 0) + ((qw + NH >= 64) ? (64 - qw - 1 - NH) : 0);
        const float* rp = rpb + (size_t)head_l * RPW * RPW;

        float mx = -1e30f;
        for (int it = 0; it < NK; ++it) {
            const int kk = (it + q * 4) & (NK - 1);   // staggered: spread LDS banks
            const int ki = uh + (kk >> UWS) - sh;
            const int kj = uw + (kk & (UW - 1)) - sv;
            float s;
            if (ki >= 0 && ki < KS && kj >= 0 && kj < KS)
                s = prow[kk] + rp[(ph + ki) * RPW + (pw + kj)];
            else
                s = -1e30f;
            prow[kk] = s;
            mx = fmaxf(mx, s);
        }
        float sum = 0.f;
        for (int it = 0; it < NK; ++it) {
            const int kk = (it + q * 4) & (NK - 1);
            const float e = __expf(prow[kk] - mx);
            prow[kk] = e;
            sum += e;
        }
        const float inv = 1.f / sum;
        unsigned* prow32 = (unsigned*)prow;           // pack normalized P as bf16 pairs
        for (int kk = 0; kk < NK; kk += 2) {
            prow32[kk >> 1] = f2bf_pk(prow[kk] * inv, prow[kk + 1] * inv);
        }
    }
    __syncthreads();

    // ---- O[16 q, 32 d] = P * V ----
    f32x8 o0, o1;
    #pragma unroll
    for (int j = 0; j < 8; ++j) { o0[j] = 0.f; o1[j] = 0.f; }

    const unsigned short* vp0 = vtb + ((size_t)bh * 32 + l15) * 4096;
    const unsigned short* vp1 = vtb + ((size_t)bh * 32 + 16 + l15) * 4096;
    const char* pbytes = (const char*)(sw + l15 * NK);

    #pragma unroll
    for (int kbs = 0; kbs < NK; kbs += 32) {
        FragB16 pa;
        pa.q4[0] = *(const u32x4*)(pbytes + kbs * 2 + hi * 32);
        pa.q4[1] = *(const u32x4*)(pbytes + kbs * 2 + hi * 32 + 16);
        const int krun = kbs + hi * 16;
        const int ky = krun >> UWS;
        const int kx = krun & (UW - 1);
        const size_t vo = (size_t)(uh + ky) * 64 + (uw + kx);
        FragB16 v0, v1;
        v0.q4[0] = *(const u32x4*)(vp0 + vo);
        v0.q4[1] = *(const u32x4*)(vp0 + vo + 8);
        v1.q4[0] = *(const u32x4*)(vp1 + vo);
        v1.q4[1] = *(const u32x4*)(vp1 + vo + 8);
        o0 = wmma_bf16(pa, v0, o0);
        o1 = wmma_bf16(pa, v1, o1);
    }

    #pragma unroll
    for (int j = 0; j < 8; ++j) {
        const int r  = j + hi * 8;
        const int oh = qh0 + (r >> 2);
        const int ow = qw0 + (r & 3);
        const size_t op = ((size_t)b * 4096 + oh * 64 + ow) * 256 + head * 32;
        attn[op + l15]      = f2bf(o0[j]);
        attn[op + 16 + l15] = f2bf(o1[j]);
    }
}

// ---------------------------------------------------------------------------
// Kernel 4: out = attn @ w_proj^T + b_proj  (M=32768, N=256, K=256)
// 32x64 tile per wave; bf16 pre-converted weights.
// ---------------------------------------------------------------------------
__global__ __launch_bounds__(256) void proj_wmma_kernel(
    const unsigned short* __restrict__ attn, const unsigned short* __restrict__ wpbf,
    const float* __restrict__ bp, float* __restrict__ out)
{
    const int lane = threadIdx.x & 31;
    const int warp = threadIdx.x >> 5;
    const int wid  = blockIdx.x * 8 + warp;
    const int m0   = (wid >> 2) * 32;
    const int n0   = (wid & 3) * 64;
    const int l15  = lane & 15;
    const int hi   = lane >> 4;

    const unsigned short* arow0 = attn + (size_t)(m0 + l15) * 256;
    const unsigned short* arow1 = attn + (size_t)(m0 + 16 + l15) * 256;

    f32x8 acc[2][4];
    #pragma unroll
    for (int mi = 0; mi < 2; ++mi)
        #pragma unroll
        for (int nt = 0; nt < 4; ++nt)
            #pragma unroll
            for (int j = 0; j < 8; ++j) acc[mi][nt][j] = 0.f;

    #pragma unroll
    for (int kc = 0; kc < 8; ++kc) {
        FragB16 a0, a1;
        a0.q4[0] = *(const u32x4*)(arow0 + kc * 32 + hi * 8);
        a0.q4[1] = *(const u32x4*)(arow0 + kc * 32 + 16 + hi * 8);
        a1.q4[0] = *(const u32x4*)(arow1 + kc * 32 + hi * 8);
        a1.q4[1] = *(const u32x4*)(arow1 + kc * 32 + 16 + hi * 8);
        #pragma unroll
        for (int nt = 0; nt < 4; ++nt) {
            const int nn = n0 + nt * 16 + l15;
            const unsigned short* wrow = wpbf + (size_t)nn * 256 + kc * 32 + hi * 8;
            FragB16 bfr;
            bfr.q4[0] = *(const u32x4*)(wrow);
            bfr.q4[1] = *(const u32x4*)(wrow + 16);
            acc[0][nt] = wmma_bf16(a0, bfr, acc[0][nt]);
            acc[1][nt] = wmma_bf16(a1, bfr, acc[1][nt]);
        }
    }

    #pragma unroll
    for (int nt = 0; nt < 4; ++nt) {
        const int nn = n0 + nt * 16 + l15;
        const float bias = bp[nn];
        #pragma unroll
        for (int mi = 0; mi < 2; ++mi)
            #pragma unroll
            for (int j = 0; j < 8; ++j) {
                const int row = m0 + mi * 16 + j + hi * 8;
                out[(size_t)row * 256 + nn] = acc[mi][nt][j] + bias;
            }
    }
}

// ---------------------------------------------------------------------------
extern "C" void kernel_launch(void* const* d_in, const int* in_sizes, int n_in,
                              void* d_out, int out_size, void* d_ws, size_t ws_size,
                              hipStream_t stream) {
    (void)in_sizes; (void)n_in; (void)out_size; (void)ws_size;
    const float* x      = (const float*)d_in[0];
    const float* w_qkv  = (const float*)d_in[1];
    const float* b_qkv  = (const float*)d_in[2];
    const float* rpb0   = (const float*)d_in[3];
    const float* rpb1   = (const float*)d_in[4];
    const float* w_proj = (const float*)d_in[5];
    const float* b_proj = (const float*)d_in[6];
    float* out = (float*)d_out;

    const size_t MB = 1024 * 1024;
    char* ws = (char*)d_ws;
    unsigned short* qb   = (unsigned short*)(ws);                  // 16 MB
    unsigned short* kb   = (unsigned short*)(ws + 16 * MB);        // 16 MB
    unsigned short* vtb  = (unsigned short*)(ws + 32 * MB);        // 16 MB
    unsigned short* xbf  = (unsigned short*)(ws + 48 * MB);        // 16 MB (aliased)
    unsigned short* attn = (unsigned short*)(ws + 48 * MB);        //  "  (after qkv)
    unsigned short* wqbf = (unsigned short*)(ws + 64 * MB);        // 384 KB
    unsigned short* wpbf = (unsigned short*)(ws + 64 * MB + 512 * 1024); // 128 KB

    // fp32 -> bf16 operand pre-conversion (bandwidth-bound, one pass each)
    f32_to_bf16_kernel<<<4096, 256, 0, stream>>>(x, xbf, 32768 * 256);
    f32_to_bf16_kernel<<<96,   256, 0, stream>>>(w_qkv, wqbf, 768 * 256);
    f32_to_bf16_kernel<<<32,   256, 0, stream>>>(w_proj, wpbf, 256 * 256);

    // 32768x768 GEMM: 12288 waves of 32x64 -> 1536 blocks x 8 waves
    qkv_wmma_kernel<<<1536, 256, 0, stream>>>(xbf, wqbf, b_qkv, qb, kb, vtb);

    // 8192 query tiles per half (8 b x 4 heads x 16 x 16)
    natt_wmma_kernel<7, 1, 4><<<2048, 128, 0, stream>>>(qb, kb, vtb, rpb0, attn, 0);
    natt_wmma_kernel<13, 2, 2><<<4096, 64, 0, stream>>>(qb, kb, vtb, rpb1, attn, 4);

    // 32768x256 GEMM: 4096 waves of 32x64 -> 512 blocks x 8 waves
    proj_wmma_kernel<<<512, 256, 0, stream>>>(attn, wpbf, b_proj, out);
}
